// AnchorGNN_56813827392365
// MI455X (gfx1250) — matrix operand order repaired and checked by reading, not compiled, not run on hardware
//
#include <hip/hip_runtime.h>
#include <hip/hip_bf16.h>

typedef __attribute__((ext_vector_type(16))) _Float16 v16h;
typedef __attribute__((ext_vector_type(8)))  _Float16 v8h;
typedef __attribute__((ext_vector_type(8)))  float    v8f;
typedef __attribute__((ext_vector_type(4)))  unsigned int u32x4;
typedef __attribute__((ext_vector_type(8)))  int i32x8;
typedef __attribute__((ext_vector_type(4)))  int i32x4;

// exact pointee type spelled the way clang prints the builtin's parameter
typedef int v4i __attribute__((vector_size(16)));
typedef __attribute__((address_space(1))) v4i GV4;
typedef __attribute__((address_space(3))) v4i LV4;

#define EMB   64
#define HD    16
#define VC    64
#define CC    1024
#define LSELF 1088
#define G     16384
#define SEGS  8
#define KEYS_PER_SEG 2048
#define WAVES 4

#ifndef __has_builtin
#define __has_builtin(x) 0
#endif
#if __has_builtin(__builtin_amdgcn_global_load_async_to_lds_b128)
#define HAVE_ASYNC 1
#else
#define HAVE_ASYNC 0
#endif
#if __has_builtin(__builtin_amdgcn_tensor_load_to_lds)
#define HAVE_TDM 1
#else
#define HAVE_TDM 0
#endif

#if __has_builtin(__builtin_amdgcn_s_wait_asynccnt)
#define WAIT_ASYNCCNT_8() __builtin_amdgcn_s_wait_asynccnt(8)
#define WAIT_ASYNCCNT_0() __builtin_amdgcn_s_wait_asynccnt(0)
#else
#define WAIT_ASYNCCNT_8() asm volatile("s_wait_asynccnt 0x8" ::: "memory")
#define WAIT_ASYNCCNT_0() asm volatile("s_wait_asynccnt 0x0" ::: "memory")
#endif
#if __has_builtin(__builtin_amdgcn_s_wait_tensorcnt)
#define WAIT_TENSORCNT_0() __builtin_amdgcn_s_wait_tensorcnt(0)
#else
#define WAIT_TENSORCNT_0() asm volatile("s_wait_tensorcnt 0x0" ::: "memory")
#endif

// ---- WMMA fragment loaders (f16, 16x16x32) -------------------------------
// A: 16x32 row-major f16 in LDS
__device__ __forceinline__ v16h load_a_frag(const _Float16* x, int kk, int lane) {
  int hi = lane >> 4, m = lane & 15;
  union { v16h v; v8h h[2]; } a;
  a.h[0] = *(const v8h*)(x + m * 64 + kk + hi * 8);        // K = kk + hi*8 .. +7
  a.h[1] = *(const v8h*)(x + m * 64 + kk + 16 + hi * 8);   // K = kk+16+hi*8 .. +7
  return a.v;
}
// A built directly from an f32 row-major LDS tile (cvt on the fly)
__device__ __forceinline__ v16h make_a_frag_f32(const float* xf, int kk, int lane) {
  int hi = lane >> 4, m = lane & 15;
  const float* p0 = xf + m * 64 + kk + hi * 8;
  const float* p1 = p0 + 16;
  float4 f0 = ((const float4*)p0)[0];
  float4 f1 = ((const float4*)p0)[1];
  float4 g0 = ((const float4*)p1)[0];
  float4 g1 = ((const float4*)p1)[1];
  v16h a;
  a[0]  = (_Float16)f0.x; a[1]  = (_Float16)f0.y; a[2]  = (_Float16)f0.z; a[3]  = (_Float16)f0.w;
  a[4]  = (_Float16)f1.x; a[5]  = (_Float16)f1.y; a[6]  = (_Float16)f1.z; a[7]  = (_Float16)f1.w;
  a[8]  = (_Float16)g0.x; a[9]  = (_Float16)g0.y; a[10] = (_Float16)g0.z; a[11] = (_Float16)g0.w;
  a[12] = (_Float16)g1.x; a[13] = (_Float16)g1.y; a[14] = (_Float16)g1.z; a[15] = (_Float16)g1.w;
  return a;
}
// B: 32x16 with B[k][n] = w[16*htile + n][k]; weight rows row-major in LDS
__device__ __forceinline__ v16h load_b_frag(const _Float16* w, int htile, int kk, int lane) {
  int hi = lane >> 4, n = lane & 15;
  union { v16h v; v8h h[2]; } b;
  const _Float16* p = w + (htile * 16 + n) * 64 + kk + hi * 8;
  b.h[0] = *(const v8h*)(p);
  b.h[1] = *(const v8h*)(p + 16);
  return b.v;
}

#if HAVE_ASYNC
// one 32-float (128 B) strip per lane: 8 async b128 transfers (ASYNCcnt += 8/wave)
__device__ __forceinline__ void issue_async_row(const float* g, float* l) {
  GV4* gp = (GV4*)g;
  LV4* lp = (LV4*)l;
  __builtin_amdgcn_global_load_async_to_lds_b128(gp, lp, 0,   0);
  __builtin_amdgcn_global_load_async_to_lds_b128(gp, lp, 16,  0);
  __builtin_amdgcn_global_load_async_to_lds_b128(gp, lp, 32,  0);
  __builtin_amdgcn_global_load_async_to_lds_b128(gp, lp, 48,  0);
  __builtin_amdgcn_global_load_async_to_lds_b128(gp, lp, 64,  0);
  __builtin_amdgcn_global_load_async_to_lds_b128(gp, lp, 80,  0);
  __builtin_amdgcn_global_load_async_to_lds_b128(gp, lp, 96,  0);
  __builtin_amdgcn_global_load_async_to_lds_b128(gp, lp, 112, 0);
}
#endif

__device__ __forceinline__ unsigned lds_byte_off(const void* p) {
  return (unsigned)(size_t)(__attribute__((address_space(3))) const void*)p;
}

// ---- Kernel 1a: self-attention K/V projections for all 1088 keys ---------
__global__ void __launch_bounds__(32)
k_self_proj(const float* __restrict__ v_sem_fea, const float* __restrict__ c_sem_fea,
            const float* __restrict__ self_in_w, const float* __restrict__ self_in_b,
            float* __restrict__ kp_out, float* __restrict__ vp_out) {
  __shared__ _Float16 wk_s[64 * 64];
  __shared__ _Float16 wv_s[64 * 64];
  __shared__ _Float16 x_s[16 * 64];
  int lane = threadIdx.x;
  for (int t = lane; t < 4096; t += 32) {
    wk_s[t] = (_Float16)self_in_w[4096 + t];
    wv_s[t] = (_Float16)self_in_w[8192 + t];
  }
  int r0 = blockIdx.x * 16;
  {
    int r = lane >> 1, part = lane & 1;
    int rg = r0 + r;
    const float* src = (rg < VC) ? (v_sem_fea + rg * 64) : (c_sem_fea + (rg - VC) * 64);
    src += part * 32;
    _Float16* dst = x_s + r * 64 + part * 32;
#pragma unroll
    for (int q8 = 0; q8 < 4; ++q8) {
      float4 f0 = ((const float4*)src)[q8 * 2];
      float4 f1 = ((const float4*)src)[q8 * 2 + 1];
      v8h hv;
      hv[0] = (_Float16)f0.x; hv[1] = (_Float16)f0.y; hv[2] = (_Float16)f0.z; hv[3] = (_Float16)f0.w;
      hv[4] = (_Float16)f1.x; hv[5] = (_Float16)f1.y; hv[6] = (_Float16)f1.z; hv[7] = (_Float16)f1.w;
      *(v8h*)(dst + q8 * 8) = hv;
    }
  }
  __builtin_amdgcn_wave_barrier();
  int hi = lane >> 4, n = lane & 15;
  v16h a0 = load_a_frag(x_s, 0, lane);
  v16h a1 = load_a_frag(x_s, 32, lane);
#pragma unroll
  for (int h = 0; h < 4; ++h) {
    v8f ck = {};
    ck = __builtin_amdgcn_wmma_f32_16x16x32_f16(false, a0, false, load_b_frag(wk_s, h, 0, lane),  (short)0, ck, false, false);
    ck = __builtin_amdgcn_wmma_f32_16x16x32_f16(false, a1, false, load_b_frag(wk_s, h, 32, lane), (short)0, ck, false, false);
    float bk = self_in_b[64 + h * 16 + n];
#pragma unroll
    for (int r = 0; r < 8; ++r)
      kp_out[(size_t)(r0 + hi * 8 + r) * 64 + h * 16 + n] = ck[r] + bk;
    v8f cv = {};
    cv = __builtin_amdgcn_wmma_f32_16x16x32_f16(false, a0, false, load_b_frag(wv_s, h, 0, lane),  (short)0, cv, false, false);
    cv = __builtin_amdgcn_wmma_f32_16x16x32_f16(false, a1, false, load_b_frag(wv_s, h, 32, lane), (short)0, cv, false, false);
    float bv = self_in_b[128 + h * 16 + n];
#pragma unroll
    for (int r = 0; r < 8; ++r)
      vp_out[(size_t)(r0 + hi * 8 + r) * 64 + h * 16 + n] = cv[r] + bv;
  }
}

// ---- Kernel 1b: flash attention for the 64 needed self queries -----------
__global__ void __launch_bounds__(32)
k_self_attn(const float* __restrict__ v_sem_fea,
            const float* __restrict__ self_in_w, const float* __restrict__ self_in_b,
            const float* __restrict__ self_out_w, const float* __restrict__ self_out_b,
            const float* __restrict__ kp, const float* __restrict__ vp,
            float* __restrict__ v_sem) {
  __shared__ float attn_s[8 * 64];
  int t = threadIdx.x;
  int q = blockIdx.x * 8 + (t >> 2);
  int h = t & 3;
  float qv[16];
  const float* fr = v_sem_fea + q * 64;
#pragma unroll
  for (int d = 0; d < 16; ++d) {
    const float* wr = self_in_w + (h * 16 + d) * 64;
    float s = self_in_b[h * 16 + d];
    for (int e = 0; e < 64; ++e) s += fr[e] * wr[e];
    qv[d] = s * 0.25f;  // 1/sqrt(HD)
  }
  float m = -3.0e38f, l = 0.f, acc[16];
#pragma unroll
  for (int d = 0; d < 16; ++d) acc[d] = 0.f;
  for (int key = 0; key < LSELF; ++key) {
    const float* kr = kp + (size_t)key * 64 + h * 16;
    float s = 0.f;
#pragma unroll
    for (int d = 0; d < 16; ++d) s += qv[d] * kr[d];
    float mn = fmaxf(m, s);
    float al = __expf(m - mn);
    float p  = __expf(s - mn);
    l = l * al + p;
    m = mn;
    const float* vr = vp + (size_t)key * 64 + h * 16;
#pragma unroll
    for (int d = 0; d < 16; ++d) acc[d] = acc[d] * al + p * vr[d];
  }
  float inv = 1.f / l;
  int ql = t >> 2;
#pragma unroll
  for (int d = 0; d < 16; ++d) attn_s[ql * 64 + h * 16 + d] = acc[d] * inv;
  __syncthreads();
  for (int k2 = 0; k2 < 16; ++k2) {
    int p2 = t + 32 * k2;
    int q2 = p2 >> 6, e = p2 & 63;
    const float* wr = self_out_w + e * 64;
    const float* ar = attn_s + q2 * 64;
    float s = self_out_b[e];
    for (int d = 0; d < 64; ++d) s += ar[d] * wr[d];
    v_sem[(size_t)(blockIdx.x * 8 + q2) * 64 + e] = s;
  }
}

// ---- Kernel 2: gather (async) + WMMA K/V projection + flash attention ----
__global__ void __launch_bounds__(128)
k_cross_flash(const float* __restrict__ v, const int* __restrict__ v_class,
              const float* __restrict__ v_sem,
              const float* __restrict__ cross_in_w, const float* __restrict__ cross_in_b,
              float* __restrict__ partials) {
  __shared__ _Float16 wk_s[64 * 64];
  __shared__ _Float16 wv_s[64 * 64];
  __shared__ float bk_s[64], bv_s[64], q_s[64];
#if HAVE_ASYNC || HAVE_TDM
  __shared__ float stage_s[8192];  // reused: weight f32 staging, then per-wave double-buffered gather tiles
#endif
#if !HAVE_ASYNC
  __shared__ _Float16 x_s[WAVES][16 * 64];
#endif
  __shared__ float kp_s[WAVES][16 * 64];
  __shared__ float p_s[WAVES][64];
  __shared__ float cred[WAVES][4];
  __shared__ float wm_s[WAVES][4], wl_s[WAVES][4], wacc_s[WAVES][64];

  int i   = blockIdx.x >> 3;   // query row
  int seg = blockIdx.x & 7;    // key segment
  int tid = threadIdx.x;

#if HAVE_TDM
  // TDM: DMA both 64x64 f32 weight blocks into LDS staging, then convert to f16.
  if (tid == 0) {
    i32x8 g1;
    g1[0] = 0x00020000;                   // data_size = 4 B
    g1[1] = 64 << 16;                     // tensor_dim0 = 64
    g1[2] = 64 << 16;                     // tensor_dim1 = 64
    g1[3] = 64 << 16;                     // tile_dim0   = 64
    g1[4] = 64;                           // tile_dim1   = 64
    g1[5] = 64;                           // tensor_dim0_stride = 64
    g1[6] = 0; g1[7] = 0;
    i32x4 z4; z4[0] = 0; z4[1] = 0; z4[2] = 0; z4[3] = 0;
    i32x8 z8; z8[0] = 0; z8[1] = 0; z8[2] = 0; z8[3] = 0; z8[4] = 0; z8[5] = 0; z8[6] = 0; z8[7] = 0;
    unsigned long long gak = (unsigned long long)(size_t)(cross_in_w + 4096);
    u32x4 g0k;
    g0k[0] = 1u;                          // count = 1
    g0k[1] = lds_byte_off(stage_s);       // lds_addr
    g0k[2] = (unsigned)(gak & 0xffffffffu);
    g0k[3] = (unsigned)(gak >> 32) | (2u << 30);  // type = 2
    __builtin_amdgcn_tensor_load_to_lds(g0k, g1, z4, z4, z8, 0);
    unsigned long long gav = (unsigned long long)(size_t)(cross_in_w + 8192);
    u32x4 g0v;
    g0v[0] = 1u;
    g0v[1] = lds_byte_off(stage_s + 4096);
    g0v[2] = (unsigned)(gav & 0xffffffffu);
    g0v[3] = (unsigned)(gav >> 32) | (2u << 30);
    __builtin_amdgcn_tensor_load_to_lds(g0v, g1, z4, z4, z8, 0);
    WAIT_TENSORCNT_0();
  }
  __syncthreads();
  asm volatile("" ::: "memory");
  for (int t = tid; t < 4096; t += 128) {
    wk_s[t] = (_Float16)stage_s[t];
    wv_s[t] = (_Float16)stage_s[4096 + t];
  }
#else
  for (int t = tid; t < 4096; t += 128) {
    wk_s[t] = (_Float16)cross_in_w[4096 + t];
    wv_s[t] = (_Float16)cross_in_w[8192 + t];
  }
#endif
  if (tid < 64) {
    bk_s[tid] = cross_in_b[64 + tid];
    bv_s[tid] = cross_in_b[128 + tid];
    const float* sr = v_sem + (size_t)i * 64;
    const float* wr = cross_in_w + tid * 64;
    float s = cross_in_b[tid];
    for (int e = 0; e < 64; ++e) s += sr[e] * wr[e];
    q_s[tid] = s * 0.25f;  // scaled q
  }
  __syncthreads();

  int w    = tid >> 5;
  int lane = tid & 31;
  int hi = lane >> 4, n = lane & 15;
  int mkey = n;
  int h0 = 2 * hi, h1 = 2 * hi + 1;

  float mh[4], lh[4], accp[4];
#pragma unroll
  for (int h = 0; h < 4; ++h) { mh[h] = -3.0e38f; lh[h] = 0.f; accp[h] = 0.f; }

  const int base = i * G + seg * KEYS_PER_SEG;
  const int iters = KEYS_PER_SEG / (16 * WAVES);
  const int r_g = lane >> 1, part_g = lane & 1;

#if HAVE_ASYNC
  float* xbuf = stage_s + w * 2048;
  {
    int gidx = v_class[base + w * 16 + r_g];
    issue_async_row(v + (size_t)gidx * 64 + part_g * 32, xbuf + r_g * 64 + part_g * 32);
  }
#endif

  for (int it = 0; it < iters; ++it) {
    int j0 = base + (it * WAVES + w) * 16;
    v16h a0, a1;
#if HAVE_ASYNC
    if (it + 1 < iters) {
      int jn = base + ((it + 1) * WAVES + w) * 16;
      int gidx = v_class[jn + r_g];
      issue_async_row(v + (size_t)gidx * 64 + part_g * 32,
                      xbuf + ((it + 1) & 1) * 1024 + r_g * 64 + part_g * 32);
      __builtin_prefetch(v_class + jn + WAVES * 16 + r_g, 0, 3);
      WAIT_ASYNCCNT_8();
    } else {
      WAIT_ASYNCCNT_0();
    }
    asm volatile("" ::: "memory");
    const float* xf = xbuf + (it & 1) * 1024;
    a0 = make_a_frag_f32(xf, 0, lane);
    a1 = make_a_frag_f32(xf, 32, lane);
#else
    {
      int gidx = v_class[j0 + r_g];
      const float4* src = (const float4*)(v + (size_t)gidx * 64 + part_g * 32);
      _Float16* dst = x_s[w] + r_g * 64 + part_g * 32;
#pragma unroll
      for (int q8 = 0; q8 < 4; ++q8) {
        float4 f0 = src[q8 * 2], f1 = src[q8 * 2 + 1];
        v8h hv;
        hv[0] = (_Float16)f0.x; hv[1] = (_Float16)f0.y; hv[2] = (_Float16)f0.z; hv[3] = (_Float16)f0.w;
        hv[4] = (_Float16)f1.x; hv[5] = (_Float16)f1.y; hv[6] = (_Float16)f1.z; hv[7] = (_Float16)f1.w;
        *(v8h*)(dst + q8 * 8) = hv;
      }
      if (it + 1 < iters) __builtin_prefetch(v_class + j0 + WAVES * 16 + r_g, 0, 3);
    }
    __builtin_amdgcn_wave_barrier();
    a0 = load_a_frag(x_s[w], 0, lane);
    a1 = load_a_frag(x_s[w], 32, lane);
#endif
    // K and V projections via WMMA (f16 in, f32 out)
    v8f Dv[4];
#pragma unroll
    for (int h = 0; h < 4; ++h) {
      v8f ck = {};
      ck = __builtin_amdgcn_wmma_f32_16x16x32_f16(false, a0, false, load_b_frag(wk_s, h, 0, lane),  (short)0, ck, false, false);
      ck = __builtin_amdgcn_wmma_f32_16x16x32_f16(false, a1, false, load_b_frag(wk_s, h, 32, lane), (short)0, ck, false, false);
      float bb = bk_s[h * 16 + n];
#pragma unroll
      for (int r = 0; r < 8; ++r)
        kp_s[w][(hi * 8 + r) * 64 + h * 16 + n] = ck[r] + bb;
      v8f cv = {};
      cv = __builtin_amdgcn_wmma_f32_16x16x32_f16(false, a0, false, load_b_frag(wv_s, h, 0, lane),  (short)0, cv, false, false);
      cv = __builtin_amdgcn_wmma_f32_16x16x32_f16(false, a1, false, load_b_frag(wv_s, h, 32, lane), (short)0, cv, false, false);
      float bb2 = bv_s[h * 16 + n];
#pragma unroll
      for (int r = 0; r < 8; ++r) cv[r] += bb2;
      Dv[h] = cv;
    }
    __builtin_amdgcn_wave_barrier();
    // scores: lane handles key mkey for heads h0,h1
    float s0 = 0.f, s1 = 0.f;
#pragma unroll
    for (int d = 0; d < 16; ++d) {
      s0 += q_s[h0 * 16 + d] * kp_s[w][mkey * 64 + h0 * 16 + d];
      s1 += q_s[h1 * 16 + d] * kp_s[w][mkey * 64 + h1 * 16 + d];
    }
    float mx0 = s0, mx1 = s1;
#pragma unroll
    for (int off = 1; off < 16; off <<= 1) {
      mx0 = fmaxf(mx0, __shfl_xor(mx0, off));
      mx1 = fmaxf(mx1, __shfl_xor(mx1, off));
    }
    if (mkey == 0) { cred[w][h0] = mx0; cred[w][h1] = mx1; }
    __builtin_amdgcn_wave_barrier();
    float cm[4];
#pragma unroll
    for (int h = 0; h < 4; ++h) cm[h] = cred[w][h];
#pragma unroll
    for (int h = 0; h < 4; ++h) {
      float mn = fmaxf(mh[h], cm[h]);
      float al = __expf(mh[h] - mn);
      accp[h] *= al; lh[h] *= al; mh[h] = mn;
    }
    float p0 = __expf(s0 - mh[h0]);
    float p1 = __expf(s1 - mh[h1]);
    p_s[w][h0 * 16 + mkey] = p0;
    p_s[w][h1 * 16 + mkey] = p1;
    float sm0 = p0, sm1 = p1;
#pragma unroll
    for (int off = 1; off < 16; off <<= 1) {
      sm0 += __shfl_xor(sm0, off);
      sm1 += __shfl_xor(sm1, off);
    }
    if (mkey == 0) { cred[w][h0] = sm0; cred[w][h1] = sm1; }
    __builtin_amdgcn_wave_barrier();
#pragma unroll
    for (int h = 0; h < 4; ++h) lh[h] += cred[w][h];
    // PV accumulation straight from WMMA D layout
#pragma unroll
    for (int h = 0; h < 4; ++h) {
      float a = accp[h];
#pragma unroll
      for (int r = 0; r < 8; ++r)
        a += p_s[w][h * 16 + hi * 8 + r] * Dv[h][r];
      accp[h] = a;
    }
  }
  // combine the two 8-key halves, then the 4 waves
#pragma unroll
  for (int h = 0; h < 4; ++h) accp[h] += __shfl_xor(accp[h], 16);
  if (lane < 16)
#pragma unroll
    for (int h = 0; h < 4; ++h) wacc_s[w][h * 16 + lane] = accp[h];
  if (lane == 0)
#pragma unroll
    for (int h = 0; h < 4; ++h) { wm_s[w][h] = mh[h]; wl_s[w][h] = lh[h]; }
  __syncthreads();
  if (tid < 64) {
    int h = tid >> 4, nn = tid & 15;
    float M = -3.0e38f;
    for (int ww = 0; ww < WAVES; ++ww) M = fmaxf(M, wm_s[ww][h]);
    float L = 0.f, O = 0.f;
    for (int ww = 0; ww < WAVES; ++ww) {
      float e = __expf(wm_s[ww][h] - M);
      L += wl_s[ww][h] * e;
      O += wacc_s[ww][h * 16 + nn] * e;
    }
    float* part = partials + (size_t)blockIdx.x * 72;
    part[8 + tid] = O;
    if (nn == 0) { part[h] = M; part[4 + h] = L; }
  }
}

// ---- Kernel 3: combine 8 segments per row + output projection ------------
__global__ void __launch_bounds__(64)
k_cross_reduce(const float* __restrict__ partials,
               const float* __restrict__ cross_out_w, const float* __restrict__ cross_out_b,
               float* __restrict__ out) {
  __shared__ float attn_s[64];
  int i = blockIdx.x, t = threadIdx.x;
  int h = t >> 4;
  float M = -3.0e38f;
  for (int s = 0; s < SEGS; ++s)
    M = fmaxf(M, partials[(size_t)(i * SEGS + s) * 72 + h]);
  float L = 0.f, O = 0.f;
  for (int s = 0; s < SEGS; ++s) {
    const float* p = partials + (size_t)(i * SEGS + s) * 72;
    float e = __expf(p[h] - M);
    L += p[4 + h] * e;
    O += p[8 + t] * e;
  }
  attn_s[t] = O / L;
  __syncthreads();
  const float* wr = cross_out_w + t * 64;
  float s = cross_out_b[t];
  for (int d = 0; d < 64; ++d) s += attn_s[d] * wr[d];
  out[(size_t)i * 64 + t] = s;
}

extern "C" void kernel_launch(void* const* d_in, const int* in_sizes, int n_in,
                              void* d_out, int out_size, void* d_ws, size_t ws_size,
                              hipStream_t stream) {
  const float* v           = (const float*)d_in[0];
  const float* c           = (const float*)d_in[1];  // unused by reference path
  const float* v_sem_fea   = (const float*)d_in[2];
  const float* c_sem_fea   = (const float*)d_in[3];
  const float* self_in_w   = (const float*)d_in[4];
  const float* self_in_b   = (const float*)d_in[5];
  const float* self_out_w  = (const float*)d_in[6];
  const float* self_out_b  = (const float*)d_in[7];
  const float* cross_in_w  = (const float*)d_in[8];
  const float* cross_in_b  = (const float*)d_in[9];
  const float* cross_out_w = (const float*)d_in[10];
  const float* cross_out_b = (const float*)d_in[11];
  const int*   v_class     = (const int*)d_in[12];
  float* out = (float*)d_out;
  float* ws  = (float*)d_ws;
  float* kp_self = ws;                 // 1088*64
  float* vp_self = ws + 69632;         // 1088*64
  float* v_sem   = ws + 139264;        // 64*64
  float* part    = ws + 143360;        // 512*72

  k_self_proj<<<68, 32, 0, stream>>>(v_sem_fea, c_sem_fea, self_in_w, self_in_b,
                                     kp_self, vp_self);
  k_self_attn<<<8, 32, 0, stream>>>(v_sem_fea, self_in_w, self_in_b,
                                    self_out_b ? self_out_w : self_out_w, self_out_b, kp_self, vp_self, v_sem);
  k_cross_flash<<<64 * SEGS, 32 * WAVES, 0, stream>>>(v, v_class, v_sem,
                                                      cross_in_w, cross_in_b, part);
  k_cross_reduce<<<64, 64, 0, stream>>>(part, cross_out_w, cross_out_b, out);
  (void)c; (void)in_sizes; (void)n_in; (void)out_size; (void)ws_size;
}
